// LowRankSVDBlock_86423331930456
// MI455X (gfx1250) — matrix operand order repaired and checked by reading, not compiled, not run on hardware
//
#include <hip/hip_runtime.h>
#include <math.h>

// ---------------------------------------------------------------------------
// Types / helpers
// ---------------------------------------------------------------------------
typedef __bf16 bf16;
typedef __attribute__((ext_vector_type(16))) __bf16 v16bf;
typedef __attribute__((ext_vector_type(8))) float v8f;

#define DEV static __device__ __forceinline__

DEV bf16 f2bf(float f) {
  unsigned u = __float_as_uint(f);
  u += 0x7FFFu + ((u >> 16) & 1u);  // round-to-nearest-even
  unsigned short s = (unsigned short)(u >> 16);
  bf16 r;
  __builtin_memcpy(&r, &s, 2);
  return r;
}

union V16U { unsigned u[8]; uint4 q[2]; v16bf v; };

// CDNA5 async global->LDS staging (ASYNCcnt), 16B per lane.
DEV void async_b128(unsigned lds_off, const void* gaddr) {
  asm volatile("global_load_async_to_lds_b128 %0, %1, off" ::"v"(lds_off),
               "v"(gaddr)
               : "memory");
}
DEV void wait_async0() { asm volatile("s_wait_asynccnt 0" ::: "memory"); }

// A fragment: 16x32 bf16, M per-lane, K striped (ISA 7.12.2).
// Lane's data = two contiguous 8-element runs -> 2x b128 loads.
DEV v16bf ld_a(const bf16* s, int ld) {
  const int lane = threadIdx.x & 31;
  const int m = lane & 15;
  const int kb = (lane < 16) ? 0 : 8;
  const bf16* row = s + m * ld + kb;
  V16U r;
  r.q[0] = *(const uint4*)(row);       // K = kb+0 .. kb+7
  r.q[1] = *(const uint4*)(row + 16);  // K = kb+16 .. kb+23
  return r.v;
}

// B fragment (32x16) from transposed storage s[n*ld + k] (K^T case):
// lane's 16 K-elements are two contiguous 8-element runs -> 2x b128 loads.
DEV v16bf ld_b_tr(const bf16* s, int ld) {
  const int lane = threadIdx.x & 31;
  const int n = lane & 15;
  const int kb = (lane < 16) ? 0 : 8;
  const bf16* row = s + n * ld + kb;
  V16U r;
  r.q[0] = *(const uint4*)(row);       // K = kb+0 .. kb+7
  r.q[1] = *(const uint4*)(row + 16);  // K = kb+16 .. kb+23
  return r.v;
}

// B fragment (32x16) from pair-packed storage: p[k/2][n] = (B[k][n],B[k+1][n])
DEV v16bf ld_b_pk(const unsigned* p, int ldp) {
  const int lane = threadIdx.x & 31;
  const int n = lane & 15;
  const int k2 = (lane < 16) ? 0 : 4;
  V16U r;
#pragma unroll
  for (int i = 0; i < 8; ++i) r.u[i] = p[(k2 + i + (i & 4)) * ldp + n];
  return r.v;
}

// interleave two bf16-pair words from adjacent K rows into packed columns
#define SEL_LO 0x05040100u  // col n   : {row0.lo16, row1.lo16}
#define SEL_HI 0x07060302u  // col n+1 : {row0.hi16, row1.hi16}

DEV v8f wmma_bf(v16bf a, v16bf b, v8f c) {
  return __builtin_amdgcn_wmma_f32_16x16x32_bf16(false, a, false, b, (short)0, c,
                                                 false, false);
}

enum { OUT_F32 = 1, ACT_GELU = 2 };

// ---------------------------------------------------------------------------
// Generic bf16 WMMA GEMM: C[z] = act(A[z] @ B[z] + bias[z]) + resid[z]
// Block tile BMxBN, K-tile 32, 256 threads = 8 waves (4x2); each wave owns a
// (BM/4)x(BN/2) C tile -> MA x NB accumulators. Double-buffered LDS pipeline:
// tile i+1 staged (A via async-to-LDS, B via v_perm pair-packing) while tile i
// computes; one barrier per K-tile. M%BM==0, N%BN==0, K%32==0 (true here).
// ---------------------------------------------------------------------------
template <int BM, int BN>
__global__ __launch_bounds__(256) void gemm_bf16_kernel(
    const bf16* __restrict__ A, int lda, long long strideA,
    const bf16* __restrict__ Bm, int ldb, long long strideB,
    void* __restrict__ C, int ldc, long long strideC,
    const float* __restrict__ bias, long long strideBias,
    const float* __restrict__ resid, int ldr, long long strideR,
    int K, int flags) {
  constexpr int MA = BM / 64;   // A fragments per wave (16 rows each)
  constexpr int NB = BN / 32;   // B fragments per wave (16 cols each)
  constexpr int LDB = BN + 8;   // packed-pair row stride (u32 units)
  __shared__ bf16 As[2 * BM * 48];
  __shared__ unsigned Bsp[2 * 16 * LDB];

  const int tid = threadIdx.x;
  const int lane = tid & 31;
  const int wave = tid >> 5;
  const int wm = (wave >> 1) * (BM / 4);
  const int wn = (wave & 1) * (BN / 2);

  const long long z = blockIdx.z;
  A += z * strideA;
  Bm += z * strideB;
  const long long m0 = (long long)blockIdx.y * BM;
  const long long n0 = (long long)blockIdx.x * BN;

  v8f acc[MA][NB] = {};

  const int arow = tid >> 2, acol = (tid & 3) * 8;  // 64 rows x 4 segs of 8
  const int kp = tid >> 4;                          // 0..15 k-pair
  const int bn = (tid & 15) * 4;                    // 0..60 col
  const unsigned a_base =
      (unsigned)(size_t)As + (unsigned)(arow * 48 + acol) * 2;

  auto stage = [&](int kt, int buf) {
    // A tile: async global->LDS (ASYNCcnt path), BM/64 x b128 per thread
    const unsigned al = a_base + (unsigned)buf * (BM * 96);
    const bf16* ag = A + (m0 + arow) * (long long)lda + kt + acol;
#pragma unroll
    for (int rr = 0; rr < BM; rr += 64)
      async_b128(al + rr * 96, ag + (long long)rr * lda);
    // B tile: two adjacent K rows -> perm-interleave into (k,k+1) pair words
    unsigned* bb = Bsp + buf * (16 * LDB) + kp * LDB + bn;
    const bf16* bg = Bm + (long long)(kt + 2 * kp) * ldb + n0 + bn;
#pragma unroll
    for (int cc = 0; cc < BN; cc += 64) {
      const uint2 r0 = *(const uint2*)(bg + cc);
      const uint2 r1 = *(const uint2*)(bg + cc + ldb);
      uint4 o;
      o.x = __builtin_amdgcn_perm(r1.x, r0.x, SEL_LO);
      o.y = __builtin_amdgcn_perm(r1.x, r0.x, SEL_HI);
      o.z = __builtin_amdgcn_perm(r1.y, r0.y, SEL_LO);
      o.w = __builtin_amdgcn_perm(r1.y, r0.y, SEL_HI);
      *(uint4*)(bb + cc) = o;
    }
    if (kt + 32 < K) __builtin_prefetch(bg + 32LL * ldb, 0, 1);
  };

  stage(0, 0);
  wait_async0();
  __syncthreads();

  for (int kt = 0; kt < K; kt += 32) {
    const int buf = (kt >> 5) & 1;
    if (kt + 32 < K) stage(kt + 32, buf ^ 1);  // overlap with compute

    const bf16* Ab = As + buf * (BM * 48);
    const unsigned* Bb = Bsp + buf * (16 * LDB);
    v16bf af[MA];
#pragma unroll
    for (int i = 0; i < MA; ++i) af[i] = ld_a(Ab + (wm + i * 16) * 48, 48);
#pragma unroll
    for (int j = 0; j < NB; ++j) {
      const v16bf bfr = ld_b_pk(Bb + wn + j * 16, LDB);
#pragma unroll
      for (int i = 0; i < MA; ++i) acc[i][j] = wmma_bf(af[i], bfr, acc[i][j]);
    }
    wait_async0();
    __syncthreads();
  }

  const int n_in = lane & 15;
  const int mbase = wm + ((lane < 16) ? 0 : 8);
  float* Cf = (float*)C + z * strideC;
  bf16* Cb = (bf16*)C + z * strideC;
  const float* biasp = bias ? bias + z * strideBias : nullptr;
  const float* resp = resid ? resid + z * strideR : nullptr;

#pragma unroll
  for (int i = 0; i < MA; ++i)
#pragma unroll
    for (int j = 0; j < NB; ++j) {
      const long long gn = n0 + wn + j * 16 + n_in;
#pragma unroll
      for (int r = 0; r < 8; ++r) {
        const long long gm = m0 + mbase + i * 16 + r;
        float v = acc[i][j][r];
        if (biasp) v += biasp[gn];
        if (flags & ACT_GELU) v = 0.5f * v * (1.0f + erff(v * 0.70710678118f));
        if (resp) v += resp[gm * (long long)ldr + gn];
        if (flags & OUT_F32) Cf[gm * (long long)ldc + gn] = v;
        else                 Cb[gm * (long long)ldc + gn] = f2bf(v);
      }
    }
}

// ---------------------------------------------------------------------------
// LayerNorm over D=1024, one block (256 thr) per row, fp32 in -> bf16 out
// ---------------------------------------------------------------------------
__global__ __launch_bounds__(256) void layernorm_kernel(
    const float* __restrict__ X, const float* __restrict__ g,
    const float* __restrict__ b, bf16* __restrict__ out, int D) {
  __shared__ float s1[256], s2[256];
  const int row = blockIdx.x;
  const int tid = threadIdx.x;
  const float4 v = ((const float4*)(X + (long long)row * D))[tid];
  s1[tid] = v.x + v.y + v.z + v.w;
  s2[tid] = v.x * v.x + v.y * v.y + v.z * v.z + v.w * v.w;
  __syncthreads();
#pragma unroll
  for (int o = 128; o > 0; o >>= 1) {
    if (tid < o) { s1[tid] += s1[tid + o]; s2[tid] += s2[tid + o]; }
    __syncthreads();
  }
  const float mean = s1[0] / (float)D;
  const float var = s2[0] / (float)D - mean * mean;
  const float rstd = rsqrtf(var + 1e-5f);
  bf16* orow = out + (long long)row * D;
  const float xv[4] = {v.x, v.y, v.z, v.w};
#pragma unroll
  for (int j = 0; j < 4; ++j) {
    const int c = tid * 4 + j;
    orow[c] = f2bf((xv[j] - mean) * rstd * g[c] + b[c]);
  }
}

__global__ void cvt_bf16_kernel(const float* __restrict__ in,
                                bf16* __restrict__ out, int n) {
  const int i = blockIdx.x * 256 + threadIdx.x;
  if (i < n) out[i] = f2bf(in[i]);
}

// ---------------------------------------------------------------------------
// Causal flash attention. Q/K/V/Y: [B*S, H*64] bf16 (ld = H*64).
// 128 threads = 4 waves, each wave owns 16 query rows; K/V tiles of 64 keys.
// K staged via async-to-LDS (read transposed for QK^T); V pair-packed via
// v_perm for the PV WMMA B operand. grid = (S/64, B*H)
// ---------------------------------------------------------------------------
__global__ __launch_bounds__(128) void attn_kernel(
    const bf16* __restrict__ Q, const bf16* __restrict__ K,
    const bf16* __restrict__ V, bf16* __restrict__ Y, int S, int H, int ld) {
  __shared__ bf16 Kls[64 * 80];      // 64 keys x 64 dh (+16 pad)
  __shared__ unsigned Vp[32 * 72];   // 32 key-pairs x 64 dh (+8 pad)
  __shared__ bf16 Pls[4 * 16 * 80];  // per-wave P tiles

  const int tid = threadIdx.x;
  const int lane = tid & 31;
  const int wave = tid >> 5;
  const int qtile = blockIdx.x;
  const int b = blockIdx.y / H, h = blockIdx.y % H;
  const long long base = (long long)b * S * ld + h * 64;
  const int qr0 = qtile * 64 + wave * 16;

  // Q fragments (A layout) straight from global: 2x b128 per 32-wide chunk
  v16bf qf0, qf1;
  {
    const int m = lane & 15;
    const int kb = (lane < 16) ? 0 : 8;
    const bf16* qrow = Q + base + (long long)(qr0 + m) * ld + kb;
    V16U r0, r1;
    r0.q[0] = *(const uint4*)(qrow);
    r0.q[1] = *(const uint4*)(qrow + 16);
    r1.q[0] = *(const uint4*)(qrow + 32);
    r1.q[1] = *(const uint4*)(qrow + 48);
    qf0 = r0.v;
    qf1 = r1.v;
  }

  float rmax[8], rsum[8];
  v8f o[4] = {};
#pragma unroll
  for (int r = 0; r < 8; ++r) { rmax[r] = -INFINITY; rsum[r] = 0.0f; }

  const int n_in = lane & 15;
  const int moff = (lane < 16) ? 0 : 8;
  const int lrow = tid >> 1;        // 0..63  (K staging)
  const int lseg = (tid & 1) * 32;  // 0 / 32
  const int vtp = tid >> 2;         // 0..31  (V key-pair)
  const int vc0 = (tid & 3) * 16;   // 0,16,32,48

  for (int kt0 = 0; kt0 <= qtile * 64; kt0 += 64) {
    // ---- stage K tile via async copy
    {
      const bf16* kg = K + base + (long long)(kt0 + lrow) * ld + lseg;
      const unsigned kd =
          (unsigned)(size_t)Kls + (unsigned)(lrow * 80 + lseg) * 2;
#pragma unroll
      for (int j = 0; j < 4; ++j) async_b128(kd + 16 * j, kg + 8 * j);
    }
    // ---- stage V tile pair-packed: Vp[t/2][e] = (V[t][e], V[t+1][e])
    {
      const bf16* vg0 = V + base + (long long)(kt0 + 2 * vtp) * ld + vc0;
      const bf16* vg1 = vg0 + ld;
      unsigned* vd = Vp + vtp * 72 + vc0;
#pragma unroll
      for (int half = 0; half < 2; ++half) {
        const uint4 a = *(const uint4*)(vg0 + 8 * half);
        const uint4 c = *(const uint4*)(vg1 + 8 * half);
        uint4 p0, p1;
        p0.x = __builtin_amdgcn_perm(c.x, a.x, SEL_LO);
        p0.y = __builtin_amdgcn_perm(c.x, a.x, SEL_HI);
        p0.z = __builtin_amdgcn_perm(c.y, a.y, SEL_LO);
        p0.w = __builtin_amdgcn_perm(c.y, a.y, SEL_HI);
        p1.x = __builtin_amdgcn_perm(c.z, a.z, SEL_LO);
        p1.y = __builtin_amdgcn_perm(c.z, a.z, SEL_HI);
        p1.z = __builtin_amdgcn_perm(c.w, a.w, SEL_LO);
        p1.w = __builtin_amdgcn_perm(c.w, a.w, SEL_HI);
        *(uint4*)(vd + 8 * half) = p0;
        *(uint4*)(vd + 8 * half + 4) = p1;
      }
    }
    wait_async0();
    __syncthreads();

    // ---- S = Q K^T (4 key sub-tiles x 2 WMMA over dh=64)
    v8f sc[4];
#pragma unroll
    for (int ts = 0; ts < 4; ++ts) {
      v8f a = {};
      a = wmma_bf(qf0, ld_b_tr(Kls + ts * 16 * 80, 80), a);
      a = wmma_bf(qf1, ld_b_tr(Kls + ts * 16 * 80 + 32, 80), a);
      sc[ts] = a;
    }

    // ---- scale + causal mask
#pragma unroll
    for (int ts = 0; ts < 4; ++ts) {
      const int t = kt0 + ts * 16 + n_in;
#pragma unroll
      for (int r = 0; r < 8; ++r) {
        const int grow = qr0 + moff + r;
        const float v = sc[ts][r] * 0.125f;  // 1/sqrt(64)
        sc[ts][r] = (t <= grow) ? v : -INFINITY;
      }
    }

    // ---- online softmax (row r lives in accumulator reg r, 16 lanes/row)
    float alpha[8];
#pragma unroll
    for (int r = 0; r < 8; ++r) {
      float tm = fmaxf(fmaxf(sc[0][r], sc[1][r]), fmaxf(sc[2][r], sc[3][r]));
#pragma unroll
      for (int m = 8; m >= 1; m >>= 1) tm = fmaxf(tm, __shfl_xor(tm, m, 32));
      const float nm = fmaxf(rmax[r], tm);
      alpha[r] = expf(rmax[r] - nm);
      rmax[r] = nm;
      float ps = 0.0f;
#pragma unroll
      for (int ts = 0; ts < 4; ++ts) {
        const float p = expf(sc[ts][r] - nm);
        sc[ts][r] = p;
        ps += p;
      }
#pragma unroll
      for (int m = 8; m >= 1; m >>= 1) ps += __shfl_xor(ps, m, 32);
      rsum[r] = rsum[r] * alpha[r] + ps;
    }
#pragma unroll
    for (int e = 0; e < 4; ++e)
#pragma unroll
      for (int r = 0; r < 8; ++r) o[e][r] *= alpha[r];

    // ---- P -> LDS (C-layout -> A-layout transpose through LDS)
    bf16* pw = Pls + wave * 16 * 80;
#pragma unroll
    for (int ts = 0; ts < 4; ++ts)
#pragma unroll
      for (int r = 0; r < 8; ++r)
        pw[(moff + r) * 80 + ts * 16 + n_in] = f2bf(sc[ts][r]);
    __syncthreads();

    // ---- O += P @ V  (2 key-chunks of 32 x 4 dh sub-tiles)
#pragma unroll
    for (int tc = 0; tc < 2; ++tc) {
      const v16bf pa = ld_a(pw + tc * 32, 80);
#pragma unroll
      for (int e = 0; e < 4; ++e)
        o[e] = wmma_bf(pa, ld_b_pk(Vp + tc * 16 * 72 + e * 16, 72), o[e]);
    }
    __syncthreads();
  }

  // ---- normalize + store
#pragma unroll
  for (int e = 0; e < 4; ++e)
#pragma unroll
    for (int r = 0; r < 8; ++r) {
      const long long grow = qr0 + moff + r;
      Y[base + grow * ld + e * 16 + n_in] = f2bf(o[e][r] / rsum[r]);
    }
}

// ---------------------------------------------------------------------------
// Host orchestration
// ---------------------------------------------------------------------------
template <int BM, int BN>
static void launch_gemm(const bf16* A, int lda, long long sA, const bf16* Bm,
                        int ldb, long long sB, void* C, int ldc, long long sC,
                        const float* bias, long long sBias, const float* resid,
                        int ldr, long long sR, int Mm, int Nn, int Kk, int nz,
                        int flags, hipStream_t stream) {
  dim3 grid(Nn / BN, Mm / BM, nz);
  gemm_bf16_kernel<BM, BN><<<grid, dim3(256), 0, stream>>>(
      A, lda, sA, Bm, ldb, sB, C, ldc, sC, bias, sBias, resid, ldr, sR, Kk,
      flags);
}

extern "C" void kernel_launch(void* const* d_in, const int* in_sizes, int n_in,
                              void* d_out, int out_size, void* d_ws,
                              size_t ws_size, hipStream_t stream) {
  (void)in_sizes; (void)n_in; (void)out_size; (void)ws_size;
  constexpr int B = 2, S = 2048, D = 1024, H = 16, DH = 64;
  constexpr int R_ATTN = 32, I = 4096;
  constexpr int M = B * S;  // 4096 token rows

  const float* hidden = (const float*)d_in[0];
  const float* ln1_g = (const float*)d_in[1];
  const float* ln1_b = (const float*)d_in[2];
  const float* ln2_g = (const float*)d_in[3];
  const float* ln2_b = (const float*)d_in[4];
  const float* q_U = (const float*)d_in[5];
  const float* q_V = (const float*)d_in[6];
  const float* q_b = (const float*)d_in[7];
  const float* k_U = (const float*)d_in[8];
  const float* k_V = (const float*)d_in[9];
  const float* k_b = (const float*)d_in[10];
  const float* v_U = (const float*)d_in[11];
  const float* v_V = (const float*)d_in[12];
  const float* v_b = (const float*)d_in[13];
  const float* out_U = (const float*)d_in[14];
  const float* out_V = (const float*)d_in[15];
  const float* out_b = (const float*)d_in[16];
  const float* fc1_U = (const float*)d_in[17];
  const float* fc1_V = (const float*)d_in[18];
  const float* fc1_b = (const float*)d_in[19];
  const float* fc2_U = (const float*)d_in[20];
  const float* fc2_V = (const float*)d_in[21];
  const float* fc2_b = (const float*)d_in[22];

  size_t off = 0;
  auto carve = [&](size_t elems, size_t esz) -> void* {
    void* p = (char*)d_ws + off;
    off += (elems * esz + 255) & ~(size_t)255;
    return p;
  };
  bf16* x_bf = (bf16*)carve((size_t)M * D, 2);
  bf16* Tq = (bf16*)carve((size_t)M * 512, 2);
  bf16* Tk = (bf16*)carve((size_t)M * 512, 2);
  bf16* Tv = (bf16*)carve((size_t)M * 512, 2);
  bf16* Qb = (bf16*)carve((size_t)M * D, 2);
  bf16* Kb = (bf16*)carve((size_t)M * D, 2);
  bf16* Vb = (bf16*)carve((size_t)M * D, 2);
  bf16* Yb = (bf16*)carve((size_t)M * D, 2);
  bf16* T_out = (bf16*)carve((size_t)M * 512, 2);
  float* hres = (float*)carve((size_t)M * D, 4);
  bf16* z_bf = (bf16*)carve((size_t)M * D, 2);
  bf16* T_fc1 = (bf16*)carve((size_t)M * 512, 2);
  bf16* h1 = (bf16*)carve((size_t)M * I, 2);
  bf16* T_fc2 = (bf16*)carve((size_t)M * 512, 2);
  bf16* qU_bf = (bf16*)carve((size_t)D * 512, 2);
  bf16* qV_bf = (bf16*)carve((size_t)H * R_ATTN * DH, 2);
  bf16* kU_bf = (bf16*)carve((size_t)D * 512, 2);
  bf16* kV_bf = (bf16*)carve((size_t)H * R_ATTN * DH, 2);
  bf16* vU_bf = (bf16*)carve((size_t)D * 512, 2);
  bf16* vV_bf = (bf16*)carve((size_t)H * R_ATTN * DH, 2);
  bf16* oU_bf = (bf16*)carve((size_t)D * 512, 2);
  bf16* oV_bf = (bf16*)carve((size_t)512 * D, 2);
  bf16* f1U_bf = (bf16*)carve((size_t)D * 512, 2);
  bf16* f1V_bf = (bf16*)carve((size_t)512 * I, 2);
  bf16* f2U_bf = (bf16*)carve((size_t)I * 512, 2);
  bf16* f2V_bf = (bf16*)carve((size_t)512 * D, 2);

  auto cvt = [&](const float* src, bf16* dst, int n) {
    cvt_bf16_kernel<<<dim3((n + 255) / 256), dim3(256), 0, stream>>>(src, dst, n);
  };

  // weight conversions to bf16
  cvt(q_U, qU_bf, D * 512);    cvt(q_V, qV_bf, H * R_ATTN * DH);
  cvt(k_U, kU_bf, D * 512);    cvt(k_V, kV_bf, H * R_ATTN * DH);
  cvt(v_U, vU_bf, D * 512);    cvt(v_V, vV_bf, H * R_ATTN * DH);
  cvt(out_U, oU_bf, D * 512);  cvt(out_V, oV_bf, 512 * D);
  cvt(fc1_U, f1U_bf, D * 512); cvt(fc1_V, f1V_bf, 512 * I);
  cvt(fc2_U, f2U_bf, I * 512); cvt(fc2_V, f2V_bf, 512 * D);

  // LN1
  layernorm_kernel<<<dim3(M), dim3(256), 0, stream>>>(hidden, ln1_g, ln1_b, x_bf, D);

  // QKV stage 1: [4096,1024] @ [1024,512] -> bf16  (128x128 tiles)
  launch_gemm<128, 128>(x_bf, D, 0, qU_bf, 512, 0, Tq, 512, 0, nullptr, 0,
                        nullptr, 0, 0, M, 512, D, 1, 0, stream);
  launch_gemm<128, 128>(x_bf, D, 0, kU_bf, 512, 0, Tk, 512, 0, nullptr, 0,
                        nullptr, 0, 0, M, 512, D, 1, 0, stream);
  launch_gemm<128, 128>(x_bf, D, 0, vU_bf, 512, 0, Tv, 512, 0, nullptr, 0,
                        nullptr, 0, 0, M, 512, D, 1, 0, stream);

  // QKV stage 2 (batched over 16 heads): [4096,32] @ [32,64] + bias (64x64)
  launch_gemm<64, 64>(Tq, 512, R_ATTN, qV_bf, DH, (long long)R_ATTN * DH, Qb, D,
                      DH, q_b, DH, nullptr, 0, 0, M, DH, R_ATTN, H, 0, stream);
  launch_gemm<64, 64>(Tk, 512, R_ATTN, kV_bf, DH, (long long)R_ATTN * DH, Kb, D,
                      DH, k_b, DH, nullptr, 0, 0, M, DH, R_ATTN, H, 0, stream);
  launch_gemm<64, 64>(Tv, 512, R_ATTN, vV_bf, DH, (long long)R_ATTN * DH, Vb, D,
                      DH, v_b, DH, nullptr, 0, 0, M, DH, R_ATTN, H, 0, stream);

  // causal flash attention
  attn_kernel<<<dim3(S / 64, B * H), dim3(128), 0, stream>>>(Qb, Kb, Vb, Yb, S, H, D);

  // out-proj: (Y @ out_U) @ out_V + out_b + hidden -> hres (fp32)
  launch_gemm<128, 128>(Yb, D, 0, oU_bf, 512, 0, T_out, 512, 0, nullptr, 0,
                        nullptr, 0, 0, M, 512, D, 1, 0, stream);
  launch_gemm<128, 128>(T_out, 512, 0, oV_bf, D, 0, hres, D, 0, out_b, 0,
                        hidden, D, 0, M, D, 512, 1, OUT_F32, stream);

  // LN2
  layernorm_kernel<<<dim3(M), dim3(256), 0, stream>>>(hres, ln2_g, ln2_b, z_bf, D);

  // MLP: gelu((z @ fc1_U) @ fc1_V + b1) ; then (h1 @ fc2_U) @ fc2_V + b2 + hres
  launch_gemm<128, 128>(z_bf, D, 0, f1U_bf, 512, 0, T_fc1, 512, 0, nullptr, 0,
                        nullptr, 0, 0, M, 512, D, 1, 0, stream);
  launch_gemm<128, 128>(T_fc1, 512, 0, f1V_bf, I, 0, h1, I, 0, fc1_b, 0,
                        nullptr, 0, 0, M, I, 512, 1, ACT_GELU, stream);
  launch_gemm<128, 128>(h1, I, 0, f2U_bf, 512, 0, T_fc2, 512, 0, nullptr, 0,
                        nullptr, 0, 0, M, 512, I, 1, 0, stream);
  launch_gemm<128, 128>(T_fc2, 512, 0, f2V_bf, D, 0, d_out, D, 0, fc2_b, 0,
                        hres, D, 0, M, D, 512, 1, OUT_F32, stream);
}
